// OutputModule_11888469475687
// MI455X (gfx1250) — matrix operand order repaired and checked by reading, not compiled, not run on hardware
//
#include <hip/hip_runtime.h>

#define Q_DIM 2048
#define E_DIM 1024
#define NB_DIM 64
#define V_DIM 32000

typedef __bf16 bf16;
typedef bf16 v16bf __attribute__((ext_vector_type(16)));
typedef bf16 v8bf  __attribute__((ext_vector_type(8)));
typedef bf16 v4bf  __attribute__((ext_vector_type(4)));
typedef float v8f  __attribute__((ext_vector_type(8)));
typedef float v4f  __attribute__((ext_vector_type(4)));
typedef int   v4i  __attribute__((ext_vector_type(4)));

#define AS1 __attribute__((address_space(1)))
#define AS3 __attribute__((address_space(3)))

#if __has_builtin(__builtin_amdgcn_global_load_async_to_lds_b128)
#define USE_ASYNC_LDS 1
#else
#define USE_ASYNC_LDS 0
#endif

// 16-byte global -> LDS copy; async (ASYNCcnt-tracked, no VGPR bounce) when available
__device__ __forceinline__ void g2l_b128(const bf16* __restrict__ g, bf16* l) {
#if USE_ASYNC_LDS
  __builtin_amdgcn_global_load_async_to_lds_b128((AS1 v4i*)g, (AS3 v4i*)l, 0, 0);
#else
  *(v8bf*)l = *(const v8bf*)g;
#endif
}

__device__ __forceinline__ void wait_async_all() {
#if USE_ASYNC_LDS
#if __has_builtin(__builtin_amdgcn_s_wait_asynccnt)
  __builtin_amdgcn_s_wait_asynccnt(0);
#else
  asm volatile("s_wait_asynccnt 0" ::: "memory");
#endif
#endif
}

// ---------------- f32 -> bf16 bulk convert (4-wide, NT reads) ----------------
__global__ __launch_bounds__(256) void f32_to_bf16_kernel(const float* __restrict__ s,
                                                          bf16* __restrict__ d, int n4) {
  int i = blockIdx.x * 256 + threadIdx.x;
  if (i >= n4) return;
  v4f v = __builtin_nontemporal_load((const v4f*)s + i);
  v4bf o;
  o.x = (bf16)v.x; o.y = (bf16)v.y; o.z = (bf16)v.z; o.w = (bf16)v.w;
  ((v4bf*)d)[i] = o;
}

// ---------------- S2[e] = sum_n hb[n,e]^2 ----------------
__global__ __launch_bounds__(256) void s2_kernel(const float* __restrict__ h,
                                                 float* __restrict__ S2) {
  int e = blockIdx.x * 256 + threadIdx.x;
  float s = 0.f;
#pragma unroll 8
  for (int n = 0; n < NB_DIM; n++) { float v = h[n * E_DIM + e]; s += v * v; }
  S2[e] = s;
}

// ---------------- lse[q,n] = logsumexp_e(q[q,e]*hb[n,e]) ----------------
__global__ __launch_bounds__(256) void lse_kernel(const float* __restrict__ q,
                                                  const float* __restrict__ h,
                                                  float* __restrict__ lse) {
  __shared__ float qrow[E_DIM];
  const int qi = blockIdx.y;
  for (int e = threadIdx.x; e < E_DIM; e += 256) qrow[e] = q[(size_t)qi * E_DIM + e];
  __syncthreads();
  const int wid = threadIdx.x >> 5, lane = threadIdx.x & 31;
  const int n = blockIdx.x * 8 + wid;
  const float* __restrict__ hr = h + (size_t)n * E_DIM;
  float m = -__builtin_inff();
#pragma unroll 4
  for (int e = lane; e < E_DIM; e += 32) m = fmaxf(m, qrow[e] * hr[e]);
#pragma unroll
  for (int off = 16; off > 0; off >>= 1) m = fmaxf(m, __shfl_xor(m, off, 32));
  float s = 0.f;
#pragma unroll 4
  for (int e = lane; e < E_DIM; e += 32) s += __expf(qrow[e] * hr[e] - m);
#pragma unroll
  for (int off = 16; off > 0; off >>= 1) s += __shfl_xor(s, off, 32);
  if (lane == 0) lse[(size_t)qi * NB_DIM + n] = m + __logf(s);
}

// ---------------- u[q,e] = q*S2 - sum_n lse[q,n]*hb[n,e]  (bf16 out) ----------------
__global__ __launch_bounds__(256) void u_kernel(const float* __restrict__ q,
                                                const float* __restrict__ h,
                                                const float* __restrict__ S2,
                                                const float* __restrict__ lse,
                                                bf16* __restrict__ u) {
  __shared__ float l[NB_DIM];
  const int qi = blockIdx.y;
  if (threadIdx.x < NB_DIM) l[threadIdx.x] = lse[(size_t)qi * NB_DIM + threadIdx.x];
  __syncthreads();
  const int e = blockIdx.x * 256 + threadIdx.x;
  float acc = q[(size_t)qi * E_DIM + e] * S2[e];
#pragma unroll 8
  for (int n = 0; n < NB_DIM; n++) acc -= l[n] * h[n * E_DIM + e];
  u[(size_t)qi * E_DIM + e] = (bf16)acc;
}

// ---------------- WMMA bf16 GEMM:  C[M,N] = A[M,K] * Bt[N,K]^T (+ epilogue) ----------
// EPI==0: Cout = acc + bias[n]                       (f32 NT store)
// EPI==1: act  = PReLU(acc + Qres[m,n] + bias[n])    (bf16 store)
#define BM 128
#define BN 128
#define BK 128
#define LDT 136  // padded LDS row stride in bf16 (272B = 17*16B -> 16B aligned rows)

template <int EPI>
__global__ __launch_bounds__(256)
void gemm_bf16_kernel(const bf16* __restrict__ A, const bf16* __restrict__ Bt,
                      int N, int K,
                      float* __restrict__ Cout, bf16* __restrict__ ActOut,
                      const float* __restrict__ Qres, const float* __restrict__ bias,
                      const float* __restrict__ prelu) {
  __shared__ __align__(16) bf16 As[2][BM * LDT];
  __shared__ __align__(16) bf16 Bs[2][BN * LDT];

  const int tid  = threadIdx.x;
  const int wid  = tid >> 5;
  const int lane = tid & 31;
  const int hh   = lane >> 4;      // half-wave select
  const int lr   = lane & 15;
  const int wm   = wid & 1;        // 2 waves in M  -> 64 rows each
  const int wn   = wid >> 1;       // 4 waves in N  -> 32 cols each
  const int block_m = blockIdx.y * BM;
  const int block_n = blockIdx.x * BN;

  v8f acc[4][2] = {};

  // stage one 128x128 bf16 tile of A and of Bt into buffer b (async -> LDS)
  auto stage = [&](int k0, int b) {
#pragma unroll
    for (int c = tid; c < (BM * BK) / 8; c += 256) {
      const int row = c >> 4;              // BK/8 == 16 chunks per row
      const int col = (c & 15) * 8;
      g2l_b128(&A[(size_t)(block_m + row) * K + k0 + col], &As[b][row * LDT + col]);
      g2l_b128(&Bt[(size_t)(block_n + row) * K + k0 + col], &Bs[b][row * LDT + col]);
    }
  };

  // consume one staged tile: 4 kk-steps x 8 WMMAs
  auto compute = [&](const bf16* as, const bf16* bs) {
#pragma unroll
    for (int kk = 0; kk < BK; kk += 32) {
      v16bf bfr[2], afr[4];
#pragma unroll
      for (int j = 0; j < 2; j++) {
        const bf16* bp = &bs[(wn * 32 + j * 16 + lr) * LDT + kk + hh * 16];
        v8bf b0 = *(const v8bf*)bp;
        v8bf b1 = *(const v8bf*)(bp + 8);
        bfr[j] = __builtin_shufflevector(b0, b1, 0, 1, 2, 3, 4, 5, 6, 7,
                                         8, 9, 10, 11, 12, 13, 14, 15);
      }
#pragma unroll
      for (int i = 0; i < 4; i++) {
        const bf16* ap = &as[(wm * 64 + i * 16 + lr) * LDT + kk];
        v8bf a0 = *(const v8bf*)(ap + hh * 16);
        v8bf a1 = *(const v8bf*)(ap + hh * 16 + 8);
        afr[i] = __builtin_shufflevector(a0, a1, 0, 1, 2, 3, 4, 5, 6, 7,
                                         8, 9, 10, 11, 12, 13, 14, 15);
      }
#pragma unroll
      for (int i = 0; i < 4; i++)
#pragma unroll
        for (int j = 0; j < 2; j++)
          acc[i][j] = __builtin_amdgcn_wmma_f32_16x16x32_bf16(
              false, afr[i], false, bfr[j], (short)0, acc[i][j], false, false);
    }
  };

  int buf = 0;
  stage(0, 0);
  wait_async_all();
  __syncthreads();

  int k0 = 0;
  for (; k0 + BK < K; k0 += BK) {          // steady state: prefetch next, compute cur
    stage(k0 + BK, buf ^ 1);
    compute(As[buf], Bs[buf]);
    wait_async_all();
    __syncthreads();
    buf ^= 1;
  }
  compute(As[buf], Bs[buf]);               // peeled tail: no prefetch, no barrier

  // epilogue: C/D layout -> VGPR r holds M = r + 8*hh, N = lr (within 16x16 tile)
  if (EPI == 0) {
#pragma unroll
    for (int i = 0; i < 4; i++) {
#pragma unroll
      for (int j = 0; j < 2; j++) {
        const int col = block_n + wn * 32 + j * 16 + lr;
        const float b = bias[col];
#pragma unroll
        for (int r = 0; r < 8; r++) {
          const int row = block_m + wm * 64 + i * 16 + hh * 8 + r;
          __builtin_nontemporal_store(acc[i][j][r] + b, &Cout[(size_t)row * N + col]);
        }
      }
    }
  } else {
    const float pa = *prelu;
#pragma unroll
    for (int i = 0; i < 4; i++) {
#pragma unroll
      for (int j = 0; j < 2; j++) {
        const int col = block_n + wn * 32 + j * 16 + lr;
        const float b = bias[col];
#pragma unroll
        for (int r = 0; r < 8; r++) {
          const int row = block_m + wm * 64 + i * 16 + hh * 8 + r;
          float z = acc[i][j][r] + Qres[(size_t)row * N + col] + b;
          float v = fmaxf(z, 0.f) + pa * fminf(z, 0.f);
          ActOut[(size_t)row * N + col] = (bf16)v;
        }
      }
    }
  }
}

extern "C" void kernel_launch(void* const* d_in, const int* in_sizes, int n_in,
                              void* d_out, int out_size, void* d_ws, size_t ws_size,
                              hipStream_t stream) {
  const float* q  = (const float*)d_in[0];
  const float* h  = (const float*)d_in[1];
  const float* Hw = (const float*)d_in[2];
  const float* Hb = (const float*)d_in[3];
  const float* Rw = (const float*)d_in[4];
  const float* Rb = (const float*)d_in[5];
  const float* pa = (const float*)d_in[6];
  float* out = (float*)d_out;

  char* ws = (char*)d_ws;
  auto carve = [&](size_t bytes) {
    char* p = ws;
    ws += (bytes + 255) & ~(size_t)255;
    return p;
  };
  float* S2     = (float*)carve((size_t)E_DIM * 4);
  float* lse    = (float*)carve((size_t)Q_DIM * NB_DIM * 4);
  bf16*  u_bf   = (bf16*)carve((size_t)Q_DIM * E_DIM * 2);
  bf16*  act_bf = (bf16*)carve((size_t)Q_DIM * E_DIM * 2);
  bf16*  Hw_bf  = (bf16*)carve((size_t)E_DIM * E_DIM * 2);
  bf16*  Rw_bf  = (bf16*)carve((size_t)V_DIM * E_DIM * 2);

  // 1) precision conversion (Rw_bf16 = 65.5MB -> fits in 192MB L2 for the big GEMM)
  {
    int n4 = E_DIM * E_DIM / 4;
    f32_to_bf16_kernel<<<(n4 + 255) / 256, 256, 0, stream>>>(Hw, Hw_bf, n4);
  }
  {
    int n4 = V_DIM * E_DIM / 4;
    f32_to_bf16_kernel<<<(n4 + 255) / 256, 256, 0, stream>>>(Rw, Rw_bf, n4);
  }
  // 2) S2[e]
  s2_kernel<<<E_DIM / 256, 256, 0, stream>>>(h, S2);
  // 3) lse[q,n]
  lse_kernel<<<dim3(NB_DIM / 8, Q_DIM), 256, 0, stream>>>(q, h, lse);
  // 4) u (bf16)
  u_kernel<<<dim3(E_DIM / 256, Q_DIM), 256, 0, stream>>>(q, h, S2, lse, u_bf);
  // 5) z = q + u @ Hw^T + Hb ; act = PReLU(z)  -> bf16
  gemm_bf16_kernel<1><<<dim3(E_DIM / BN, Q_DIM / BM), 256, 0, stream>>>(
      u_bf, Hw_bf, E_DIM, E_DIM, nullptr, act_bf, q, Hb, pa);
  // 6) out = act @ Rw^T + Rb  (dominant GEMM: 2048x32000x1024)
  gemm_bf16_kernel<0><<<dim3(V_DIM / BN, Q_DIM / BM), 256, 0, stream>>>(
      act_bf, Rw_bf, V_DIM, E_DIM, out, nullptr, nullptr, Rb, nullptr);
}
